// VectorQuantizer_54296976556832
// MI455X (gfx1250) — matrix operand order repaired and checked by reading, not compiled, not run on hardware
//
#include <hip/hip_runtime.h>

// ---------------------------------------------------------------------------
// VQ-VAE vector-quantizer forward for MI455X (gfx1250, wave32, WMMA)
//   h   = x @ preW^T + pre_b                    (bf16 WMMA, f32 accum)
//   S   = h @ codebook^T                        (bf16 WMMA, f32 accum)
//   idx = argmin_k (||e_k||^2 - 2 S[n,k])
//   P   = codebook @ postW^T  (tiny GEMM), out[n,:] = P[idx_n,:] + post_b
//   loss = 1.25 * mean(||h_n - e_idx||^2)
// GEMM main loop: double-buffered LDS fed by GLOBAL_LOAD_ASYNC_TO_LDS_B128
// (ASYNCcnt) when the toolchain exposes it; sync global->VGPR->LDS otherwise.
// ---------------------------------------------------------------------------

typedef __bf16 bf16_t;
typedef __attribute__((ext_vector_type(16))) __bf16 bf16x16;
typedef __attribute__((ext_vector_type(8)))  __bf16 bf16x8;
typedef __attribute__((ext_vector_type(8)))  float  f32x8;
typedef __attribute__((ext_vector_type(4)))  int    i32x4;

static constexpr int CH    = 512;          // C (embedding dim)
static constexpr int DIM   = 4048;         // D (conv dim)
static constexpr int DIMP  = 4064;         // D padded to multiple of 32 (WMMA K)
static constexpr int NCODE = 512;          // K codebook entries
static constexpr int NROW  = 8 * 1024;     // B*T = 8192
static constexpr float COMMIT = 0.25f;

#define LDS_STR 40   // bf16 elems per LDS row: 32 data + 8 pad -> conflict-free b128

#if __has_builtin(__builtin_amdgcn_global_load_async_to_lds_b128) && \
    __has_builtin(__builtin_amdgcn_s_wait_asynccnt)
#define VQ_ASYNC_LDS 1
#else
#define VQ_ASYNC_LDS 0
#endif

#if VQ_ASYNC_LDS
typedef __attribute__((address_space(1))) i32x4 glob_i32x4;
typedef __attribute__((address_space(3))) i32x4 lds_i32x4;

__device__ __forceinline__ void async_copy_b128(const void* g, void* l) {
    // global -> LDS DMA, per-lane 16B; tracked by ASYNCcnt
    __builtin_amdgcn_global_load_async_to_lds_b128(
        (glob_i32x4*)(g), (lds_i32x4*)(l), 0, 0);
}
#endif

__device__ __forceinline__ bf16x16 frag_combine(bf16x8 lo, bf16x8 hi) {
    bf16x16 r;
#pragma unroll
    for (int i = 0; i < 8; ++i) { r[i] = lo[i]; r[i + 8] = hi[i]; }
    return r;
}

// ---------------------------------------------------------------------------
// C = A @ B^T  with A[M,K] bf16 row-major (lda), B[N,K] bf16 row-major (ldb).
// Block: 256 threads = 8 waves -> 64(M) x 128(N) tile, K stepped by 32.
// Wave (wm in 0..3, wn in 0..1) owns 16x64 = 4 WMMA accumulators.
// Cols in [Nvalid,Nout) stored as 0 (zero K-padding for the next GEMM);
// cols >= Nout not stored. B row reads clamped to nbRows-1 (safe, masked out).
// ---------------------------------------------------------------------------
template<bool OUT_BF16, bool HAS_BIAS>
__global__ __launch_bounds__(256)
void gemm_bf16_wmma(const bf16_t* __restrict__ A, int lda,
                    const bf16_t* __restrict__ B, int ldb, int nbRows,
                    void* __restrict__ Cout, int ldc,
                    const float* __restrict__ bias,
                    int Nvalid, int Nout, int K)
{
#if VQ_ASYNC_LDS
    __shared__ __align__(16) bf16_t As[2][64  * LDS_STR];
    __shared__ __align__(16) bf16_t Bs[2][128 * LDS_STR];
#else
    __shared__ __align__(16) bf16_t As1[64  * LDS_STR];
    __shared__ __align__(16) bf16_t Bs1[128 * LDS_STR];
#endif

    const int tid  = threadIdx.x;
    const int lane = tid & 31;
    const int wid  = tid >> 5;
    const int wm   = wid >> 1;          // 0..3
    const int wn   = wid & 1;           // 0..1

    const int bm = blockIdx.x * 64;
    const int bn = blockIdx.y * 128;

    f32x8 acc[4] = {};

    // cooperative tile staging: each thread moves three 16-byte chunks
    const int ldRow = tid >> 2;         // 0..63
    const int ldCol = (tid & 3) * 8;    // 0,8,16,24
    int bg0 = bn + ldRow;       if (bg0 >= nbRows) bg0 = nbRows - 1;
    int bg1 = bn + 64 + ldRow;  if (bg1 >= nbRows) bg1 = nbRows - 1;

    // fragment addressing (wave32):
    //  A 16x32: lane<16 -> K chunks [0,8)+[16,24); lane>=16 -> [8,16)+[24,32)
    //  B 32x16: lane<16 -> K chunks [0,8)+[16,24); lane>=16 -> [8,16)+[24,32)
    const int fr    = lane & 15;
    const int half  = lane >> 4;
    const int koffA = half * 8;
    const int koffB = half * 8;
    const int arow  = wm * 16 + fr;

    // load all 5 fragments, then fire the 4 WMMAs back-to-back (single dscnt wait)
    auto compute_tile = [&](const bf16_t* Ab, const bf16_t* Bb) {
        bf16x8 a0 = *(const bf16x8*)(Ab + arow * LDS_STR + koffA);
        bf16x8 a1 = *(const bf16x8*)(Ab + arow * LDS_STR + koffA + 16);
        bf16x16 afrag = frag_combine(a0, a1);
        bf16x16 bfrag[4];
#pragma unroll
        for (int t = 0; t < 4; ++t) {
            const int brow = wn * 64 + t * 16 + fr;
            bf16x8 b0 = *(const bf16x8*)(Bb + brow * LDS_STR + koffB);
            bf16x8 b1 = *(const bf16x8*)(Bb + brow * LDS_STR + koffB + 16);
            bfrag[t] = frag_combine(b0, b1);
        }
#pragma unroll
        for (int t = 0; t < 4; ++t)
            acc[t] = __builtin_amdgcn_wmma_f32_16x16x32_bf16(
                false, afrag, false, bfrag[t], (short)0, acc[t], false, false);
    };

#if VQ_ASYNC_LDS
    auto issue_tile = [&](int kt, int buf) {
        async_copy_b128(A + (size_t)(bm + ldRow) * lda + kt + ldCol,
                        &As[buf][ldRow * LDS_STR + ldCol]);
        async_copy_b128(B + (size_t)bg0 * ldb + kt + ldCol,
                        &Bs[buf][ldRow * LDS_STR + ldCol]);
        async_copy_b128(B + (size_t)bg1 * ldb + kt + ldCol,
                        &Bs[buf][(64 + ldRow) * LDS_STR + ldCol]);
    };
    issue_tile(0, 0);
    int buf = 0;
    for (int kt = 0; kt < K; kt += 32) {
        __builtin_amdgcn_s_wait_asynccnt(0);   // my tile-kt writes are in LDS
        __syncthreads();                       // everyone's are; buf^1 free to reuse
        if (kt + 32 < K) issue_tile(kt + 32, buf ^ 1);  // DMA overlaps compute
        compute_tile(As[buf], Bs[buf]);
        buf ^= 1;
    }
#else
    for (int kt = 0; kt < K; kt += 32) {
        bf16x8 av  = *(const bf16x8*)(A + (size_t)(bm + ldRow) * lda + kt + ldCol);
        bf16x8 bv0 = *(const bf16x8*)(B + (size_t)bg0 * ldb + kt + ldCol);
        bf16x8 bv1 = *(const bf16x8*)(B + (size_t)bg1 * ldb + kt + ldCol);
        if (kt + 32 < K) {  // gfx1250 global_prefetch_b8 of next K tile
            __builtin_prefetch(A + (size_t)(bm + ldRow) * lda + kt + 32 + ldCol, 0, 1);
            __builtin_prefetch(B + (size_t)bg0 * ldb + kt + 32 + ldCol, 0, 1);
        }
        __syncthreads();
        *(bf16x8*)(As1 + ldRow * LDS_STR + ldCol)        = av;
        *(bf16x8*)(Bs1 + ldRow * LDS_STR + ldCol)        = bv0;
        *(bf16x8*)(Bs1 + (64 + ldRow) * LDS_STR + ldCol) = bv1;
        __syncthreads();
        compute_tile(As1, Bs1);
    }
#endif

    // epilogue: C/D 16x16 f32 layout — lanes 0-15: N=lane, M=r; lanes 16-31: M=r+8
    const int gm    = bm + wm * 16;
    const int gn    = bn + wn * 64;
    const int rbase = half * 8;
#pragma unroll
    for (int r = 0; r < 8; ++r) {
        const int grow = gm + rbase + r;
#pragma unroll
        for (int t = 0; t < 4; ++t) {
            const int gcol = gn + t * 16 + fr;
            float v = (gcol < Nvalid)
                        ? (acc[t][r] + (HAS_BIAS ? bias[gcol] : 0.0f))
                        : 0.0f;
            if (gcol < Nout) {
                if (OUT_BF16)
                    ((bf16_t*)Cout)[(size_t)grow * ldc + gcol] = (bf16_t)v;
                else
                    ((float*)Cout)[(size_t)grow * ldc + gcol] = v;
            }
        }
    }
}

// ---------------------------------------------------------------------------
// f32 -> bf16 with optional zero K-padding (dc >= sc)
// ---------------------------------------------------------------------------
__global__ void cvt_bf16_pad(const float* __restrict__ src, bf16_t* __restrict__ dst,
                             int rows, int sc, int dc)
{
    size_t i = (size_t)blockIdx.x * blockDim.x + threadIdx.x;
    size_t total = (size_t)rows * dc;
    if (i >= total) return;
    int c = (int)(i % dc);
    int r = (int)(i / dc);
    float v = (c < sc) ? src[(size_t)r * sc + c] : 0.0f;
    dst[i] = (bf16_t)v;
}

// per-row sum of squares (deterministic fixed-order reduction)
template<typename SrcT>
__global__ __launch_bounds__(256)
void row_sumsq(const SrcT* __restrict__ src, float* __restrict__ out, int cols)
{
    __shared__ float red[256];
    const int r = blockIdx.x;
    float s = 0.0f;
    for (int c = threadIdx.x; c < cols; c += 256) {
        float v = (float)src[(size_t)r * cols + c];
        s += v * v;
    }
    red[threadIdx.x] = s;
    __syncthreads();
    for (int off = 128; off; off >>= 1) {
        if (threadIdx.x < off) red[threadIdx.x] += red[threadIdx.x + off];
        __syncthreads();
    }
    if (threadIdx.x == 0) out[r] = red[0];
}

// ---------------------------------------------------------------------------
// per row: argmin_k (cnorm[k] - 2 S[n,k]); out[n,:] = P[idx,:] + post_b
// d2[n] = ||h_n||^2 + min-score  (= ||h_n - e_idx||^2)
// ---------------------------------------------------------------------------
__global__ __launch_bounds__(512)
void vq_select_out(const float* __restrict__ S, const float* __restrict__ cnorm,
                   const float* __restrict__ rowsq, const float* __restrict__ P,
                   const float* __restrict__ postB, float* __restrict__ out,
                   float* __restrict__ d2)
{
    __shared__ float sc[512];
    __shared__ int   si[512];
    __shared__ int   bidx;
    const int n = blockIdx.x;
    const int t = threadIdx.x;

    sc[t] = cnorm[t] - 2.0f * S[(size_t)n * NCODE + t];
    si[t] = t;
    __syncthreads();
    for (int off = 256; off; off >>= 1) {
        if (t < off) {
            float a = sc[t], b = sc[t + off];
            if (b < a || (b == a && si[t + off] < si[t])) { sc[t] = b; si[t] = si[t + off]; }
        }
        __syncthreads();
    }
    if (t == 0) {
        bidx = si[0];
        d2[n] = rowsq[n] + sc[0];
    }
    __syncthreads();
    const int k = bidx;
    out[(size_t)n * CH + t] = P[(size_t)k * CH + t] + postB[t];
}

__global__ __launch_bounds__(256)
void vq_finalize(const float* __restrict__ d2, float* __restrict__ out_loss)
{
    __shared__ float red[256];
    float s = 0.0f;
    for (int i = threadIdx.x; i < NROW; i += 256) s += d2[i];
    red[threadIdx.x] = s;
    __syncthreads();
    for (int off = 128; off; off >>= 1) {
        if (threadIdx.x < off) red[threadIdx.x] += red[threadIdx.x + off];
        __syncthreads();
    }
    if (threadIdx.x == 0)
        *out_loss = (1.0f + COMMIT) * red[0] / ((float)NROW * (float)DIM);
}

// ---------------------------------------------------------------------------
extern "C" void kernel_launch(void* const* d_in, const int* in_sizes, int n_in,
                              void* d_out, int out_size, void* d_ws, size_t ws_size,
                              hipStream_t stream)
{
    (void)in_sizes; (void)n_in; (void)out_size; (void)ws_size;
    const float* x     = (const float*)d_in[0];   // [8192,512]
    const float* preW  = (const float*)d_in[1];   // [4048,512]
    const float* preB  = (const float*)d_in[2];   // [4048]
    const float* cbook = (const float*)d_in[3];   // [512,4048]
    const float* postW = (const float*)d_in[4];   // [512,4048]
    const float* postB = (const float*)d_in[5];   // [512]
    float* out = (float*)d_out;                   // 8192*512 floats + 1 loss

    char* base = (char*)d_ws;
    size_t off = 0;
    auto carve = [&](size_t bytes) -> void* {
        void* p = base + off;
        off = (off + bytes + 255) & ~(size_t)255;
        return p;
    };
    bf16_t* Hbf   = (bf16_t*)carve((size_t)NROW  * DIMP * 2);  // 66.6 MB
    bf16_t* Xbf   = (bf16_t*)carve((size_t)NROW  * CH   * 2);  //  8.4 MB
    bf16_t* Wpre  = (bf16_t*)carve((size_t)DIM   * CH   * 2);  //  4.1 MB
    bf16_t* Cbbf  = (bf16_t*)carve((size_t)NCODE * DIMP * 2);  //  4.2 MB
    bf16_t* Wpost = (bf16_t*)carve((size_t)CH    * DIMP * 2);  //  4.2 MB
    float*  Smat  = (float*) carve((size_t)NROW  * NCODE * 4); // 16.8 MB
    float*  Pmat  = (float*) carve((size_t)NCODE * CH    * 4); //  1.0 MB
    float*  rowsq = (float*) carve((size_t)NROW * 4);
    float*  cnorm = (float*) carve((size_t)NCODE * 4);
    float*  d2    = (float*) carve((size_t)NROW * 4);

    auto nb = [](size_t n) { return (unsigned)((n + 255) / 256); };

    // 1. precision conversion + zero K-padding (4048 -> 4064)
    cvt_bf16_pad<<<nb((size_t)NROW * CH),   256, 0, stream>>>(x,     Xbf,   NROW,  CH,  CH);
    cvt_bf16_pad<<<nb((size_t)DIM * CH),    256, 0, stream>>>(preW,  Wpre,  DIM,   CH,  CH);
    cvt_bf16_pad<<<nb((size_t)NCODE * DIMP),256, 0, stream>>>(cbook, Cbbf,  NCODE, DIM, DIMP);
    cvt_bf16_pad<<<nb((size_t)CH * DIMP),   256, 0, stream>>>(postW, Wpost, CH,    DIM, DIMP);
    row_sumsq<float><<<NCODE, 256, 0, stream>>>(cbook, cnorm, DIM);

    // 2. H = X @ preW^T + pre_b   (M=8192, N=4048(->4064 padded), K=512)
    gemm_bf16_wmma<true, true><<<dim3(NROW / 64, (DIMP + 127) / 128), 256, 0, stream>>>(
        Xbf, CH, Wpre, CH, DIM, Hbf, DIMP, preB, DIM, DIMP, CH);

    row_sumsq<bf16_t><<<NROW, 256, 0, stream>>>(Hbf, rowsq, DIMP);

    // 3. S = H @ codebook^T   (M=8192, N=512, K=4064)
    gemm_bf16_wmma<false, false><<<dim3(NROW / 64, NCODE / 128), 256, 0, stream>>>(
        Hbf, DIMP, Cbbf, DIMP, NCODE, Smat, NCODE, nullptr, NCODE, NCODE, DIMP);

    // 4. P = codebook @ postW^T   (M=512, N=512, K=4064)
    gemm_bf16_wmma<false, false><<<dim3(NCODE / 64, CH / 128), 256, 0, stream>>>(
        Cbbf, DIMP, Wpost, DIMP, CH, Pmat, CH, nullptr, CH, CH, DIMP);

    // 5. argmin + gather + bias; 6. loss
    vq_select_out<<<NROW, 512, 0, stream>>>(Smat, cnorm, rowsq, Pmat, postB, out, d2);
    vq_finalize<<<1, 256, 0, stream>>>(d2, out + (size_t)NROW * CH);
}
